// Model_2559800508604
// MI455X (gfx1250) — compile-verified, matrix operand order checked by reading, not run on hardware
//
#include <hip/hip_runtime.h>
#include <math.h>

typedef __attribute__((ext_vector_type(16))) _Float16 v16h;
typedef __attribute__((ext_vector_type(8)))  float    v8f;

constexpr int Bc = 4, Sc = 512, Dc = 768, Ec = 12, Kc = 8, Pc = 132;
constexpr int BPc = Bc * Pc;          // 528 = 33 * 16
constexpr int MT  = BPc / 16;         // 33 M-tiles
constexpr int NT  = Dc / 16;          // 48 N-tiles

// ---------------------------------------------------------------------------
// Kernel 1: per-entity masked logsumexp over K mention rows.
//   ent_att[b,e,s] = logsumexp_k( attention[b, pos+1, s], weights=valid )
//   ent_emb[b,e,d] = logsumexp_k( hidden[b, pos+1, d],   weights=valid )
// One block per (b,e); threads stride over S then D; per-k columns are
// coalesced across lanes and L2-resident.
// ---------------------------------------------------------------------------
__global__ void ent_kernel(const float* __restrict__ att,
                           const float* __restrict__ hid,
                           const int*   __restrict__ mpos,
                           const int*   __restrict__ mmask,
                           float* __restrict__ ent_att,
                           float* __restrict__ ent_emb) {
    const int be = blockIdx.x;          // b*E + e
    const int b  = be / Ec;

    int tok[Kc]; int val[Kc];
#pragma unroll
    for (int k = 0; k < Kc; ++k) {
        tok[k] = mpos[be * Kc + k] + 1;                // pos + 1
        val[k] = (k == 0) || (mmask[be * Kc + k] > 0); // k=0 forced valid
    }

    // attention rows -> ent_att
    for (int s = threadIdx.x; s < Sc; s += blockDim.x) {
        float xs[Kc];
        float m = -INFINITY;
#pragma unroll
        for (int k = 0; k < Kc; ++k) {
            float x = att[((size_t)(b * Sc + tok[k])) * Sc + s];
            xs[k] = x;
            if (val[k]) m = fmaxf(m, x);
        }
        float sum = 0.f;
#pragma unroll
        for (int k = 0; k < Kc; ++k)
            if (val[k]) sum += __expf(xs[k] - m);
        ent_att[(size_t)be * Sc + s] = m + __logf(sum);
    }

    // hidden rows -> ent_emb
    for (int d = threadIdx.x; d < Dc; d += blockDim.x) {
        float xs[Kc];
        float m = -INFINITY;
#pragma unroll
        for (int k = 0; k < Kc; ++k) {
            float x = hid[((size_t)(b * Sc + tok[k])) * Dc + d];
            xs[k] = x;
            if (val[k]) m = fmaxf(m, x);
        }
        float sum = 0.f;
#pragma unroll
        for (int k = 0; k < Kc; ++k)
            if (val[k]) sum += __expf(xs[k] - m);
        ent_emb[(size_t)be * Dc + d] = m + __logf(sum);
    }
}

// ---------------------------------------------------------------------------
// Kernel 2: per-pair local-context logsumexp + gathers.
// One block per pair (528 blocks, 256 threads = 8 wave32).
//   c_s = ent_att[b,h,s] * ent_att[b,t,s]      (staged in LDS)
//   lc[d] = logsumexp_s( c_s * hidden[b,s,d] ) (single-pass online update,
//                                               hidden read once, coalesced)
// Each thread owns d, d+256, d+512 (768 = 3*256). Writes slots 0..3 of out.
// ---------------------------------------------------------------------------
__global__ void pair_kernel(const float* __restrict__ hid,
                            const int*   __restrict__ pairs,
                            const float* __restrict__ ent_att,
                            const float* __restrict__ ent_emb,
                            float* __restrict__ out) {
    const int idx = blockIdx.x;          // b*P + p
    const int b   = idx / Pc;
    const int hI  = pairs[idx * 2 + 0];
    const int tI  = pairs[idx * 2 + 1];

    __shared__ float cs[Sc];
    const float* ah = ent_att + (size_t)(b * Ec + hI) * Sc;
    const float* at = ent_att + (size_t)(b * Ec + tI) * Sc;
    for (int s = threadIdx.x; s < Sc; s += blockDim.x)
        cs[s] = ah[s] * at[s];
    __syncthreads();

    const int    d0 = threadIdx.x;                 // + 256, + 512
    const float* hb = hid + (size_t)b * Sc * Dc;

    float m0 = -INFINITY, m1 = -INFINITY, m2 = -INFINITY;
    float s0 = 0.f, s1 = 0.f, s2 = 0.f;

#pragma unroll 4
    for (int s = 0; s < Sc; ++s) {
        const float  c  = cs[s];
        const float* hr = hb + (size_t)s * Dc + d0;
        if ((s & 15) == 0)                         // gfx1250 global_prefetch_b8
            __builtin_prefetch(hr + 16 * Dc, 0, 1);
        float x, nm;
        x = c * hr[0];
        nm = fmaxf(m0, x); s0 = s0 * __expf(m0 - nm) + __expf(x - nm); m0 = nm;
        x = c * hr[256];
        nm = fmaxf(m1, x); s1 = s1 * __expf(m1 - nm) + __expf(x - nm); m1 = nm;
        x = c * hr[512];
        nm = fmaxf(m2, x); s2 = s2 * __expf(m2 - nm) + __expf(x - nm); m2 = nm;
    }

    const float* eh = ent_emb + (size_t)(b * Ec + hI) * Dc;
    const float* et = ent_emb + (size_t)(b * Ec + tI) * Dc;
    float* ob = out + (size_t)idx * 5 * Dc;

    const float lc0 = m0 + __logf(s0);
    const float lc1 = m1 + __logf(s1);
    const float lc2 = m2 + __logf(s2);

    // slot 0: cls = hidden[b,0,:]; slot 1: eh; slot 2: et; slot 3: lc
    ob[0 * Dc + d0      ] = hb[d0      ];
    ob[0 * Dc + d0 + 256] = hb[d0 + 256];
    ob[0 * Dc + d0 + 512] = hb[d0 + 512];
    ob[1 * Dc + d0      ] = eh[d0      ];
    ob[1 * Dc + d0 + 256] = eh[d0 + 256];
    ob[1 * Dc + d0 + 512] = eh[d0 + 512];
    ob[2 * Dc + d0      ] = et[d0      ];
    ob[2 * Dc + d0 + 256] = et[d0 + 256];
    ob[2 * Dc + d0 + 512] = et[d0 + 512];
    ob[3 * Dc + d0      ] = lc0;
    ob[3 * Dc + d0 + 256] = lc1;
    ob[3 * Dc + d0 + 512] = lc2;
}

// ---------------------------------------------------------------------------
// Kernel 3: type-embedding GEMM on the matrix unit.
//   tvec = (onehot(th)+onehot(tt)) @ W_type + b_type   -> out slot 4
// [528,6] x [6,768] mapped to v_wmma_f32_16x16x32_f16, K padded 6 -> 32.
// Fully branchless and register-only: every lane issues unconditional,
// in-bounds loads (addresses use lane&15), the upper half-wave is zeroed by
// value-selects AFTER the loads, so there are no exec-mask branches and EXEC
// is all-1s through the WMMA.
// ISA 16-bit layouts (wave32):
//   A 16x32 : lanes 0-15 = rows M=0..15, elements 0..7 <-> K=0..7; lanes
//             16-31 / elements 8..15 carry K>=8 -> zero.
//   B 32x16 : lanes 0-15 = cols N=0..15, elements 0..7 <-> K=0..7; lanes
//             16-31 / elements 8..15 carry K>=8 -> zero.
//   C/D v8f : element r, lane l -> M = r + 8*(l>>4), N = l&15.
// One wave per 16x16 tile of [528, 768]; grid 33 x 48.
// ---------------------------------------------------------------------------
__global__ void tvec_wmma_kernel(const float* __restrict__ W_type,
                                 const float* __restrict__ b_type,
                                 const int*   __restrict__ pairs,
                                 const int*   __restrict__ types,
                                 float* __restrict__ out) {
    const int mtile = blockIdx.x;      // 0..32
    const int ntile = blockIdx.y;      // 0..47
    const int lane  = threadIdx.x;     // 0..31 (one wave32)
    const int lo16  = (lane < 16);     // active half for A rows / B cols

    // ---- A: one-hot(th) + one-hot(tt) for pair row (loads unconditional) ---
    const int row = mtile * 16 + (lane & 15);   // pair flat index, < 528
    const int b   = row / Pc;
    const int hI  = pairs[row * 2 + 0];
    const int tI  = pairs[row * 2 + 1];
    const int t1  = types[b * Ec + hI];
    const int t2  = types[b * Ec + tI];
    // mask classes (not loads): upper half-wave gets no match -> all zeros
    const int c1  = lo16 ? t1 : -1;
    const int c2  = lo16 ? t2 : -1;

    v16h A;
#pragma unroll
    for (int i = 0; i < 16; ++i)
        A[i] = (_Float16)(float)((i == c1) + (i == c2));   // one-hot sum row

    // ---- B: W_type column; unconditional clause of 6 loads, mask values ----
    const int d = ntile * 16 + (lane & 15);     // output column, < 768
    float w[6];
#pragma unroll
    for (int k = 0; k < 6; ++k)
        w[k] = W_type[k * Dc + d];              // in-bounds for every lane

    v16h Bv;
#pragma unroll
    for (int k = 0; k < 6; ++k)
        Bv[k] = (_Float16)(lo16 ? w[k] : 0.f);  // select value, not load
#pragma unroll
    for (int k = 6; k < 16; ++k)
        Bv[k] = (_Float16)0.f;

    v8f c = {};
    c = __builtin_amdgcn_wmma_f32_16x16x32_f16(
            /*neg_a=*/false, A, /*neg_b=*/false, Bv,
            /*c_mod=*/(short)0, c, /*reuse_a=*/false, /*reuse_b=*/false);

    const int n    = lane & 15;
    const int mhi  = (lane >> 4) * 8;
    const int col  = ntile * 16 + n;
    const float bt = b_type[col];
#pragma unroll
    for (int r = 0; r < 8; ++r) {
        const int orow = mtile * 16 + mhi + r;
        out[((size_t)orow * 5 + 4) * Dc + col] = c[r] + bt;
    }
}

// ---------------------------------------------------------------------------
extern "C" void kernel_launch(void* const* d_in, const int* in_sizes, int n_in,
                              void* d_out, int out_size, void* d_ws, size_t ws_size,
                              hipStream_t stream) {
    const float* hidden    = (const float*)d_in[0]; // [B,S,D]
    const float* attention = (const float*)d_in[1]; // [B,S,S]
    const float* W_type    = (const float*)d_in[2]; // [6,D]
    const float* b_type    = (const float*)d_in[3]; // [D]
    const int*   mpos      = (const int*)d_in[4];   // [B,E,K]
    const int*   mmask     = (const int*)d_in[5];   // [B,E,K]
    const int*   pairs     = (const int*)d_in[6];   // [B,P,2]
    const int*   types     = (const int*)d_in[7];   // [B,E]
    float*       out       = (float*)d_out;         // [B*P, 5, D]

    float* ent_att = (float*)d_ws;                   // B*E*S = 24576 floats
    float* ent_emb = ent_att + (size_t)Bc * Ec * Sc; // B*E*D = 36864 floats

    ent_kernel<<<Bc * Ec, 256, 0, stream>>>(attention, hidden, mpos, mmask,
                                            ent_att, ent_emb);
    pair_kernel<<<BPc, 256, 0, stream>>>(hidden, pairs, ent_att, ent_emb, out);
    tvec_wmma_kernel<<<dim3(MT, NT), 32, 0, stream>>>(W_type, b_type, pairs,
                                                      types, out);
}
